// Transformer_66340064854625
// MI455X (gfx1250) — matrix operand order repaired and checked by reading, not compiled, not run on hardware
//
#include <hip/hip_runtime.h>
#include <hip/hip_bf16.h>
#include <math.h>

// Problem constants (from reference)
#define CB 4
#define CG 64
#define CT 447
#define CD 1024
#define CH 16
#define CHK 8
#define CHD 64
#define CL 8
#define CFD 256
#define CS 512          // G + T + 1
#define CHID 2816
#define CMAXSPD 20

typedef _Float16 v16h  __attribute__((ext_vector_type(16)));
typedef float    v8f   __attribute__((ext_vector_type(8)));
typedef __fp16   fp16x2 __attribute__((ext_vector_type(2)));
typedef unsigned int v4u __attribute__((ext_vector_type(4)));
typedef int      v8i   __attribute__((ext_vector_type(8)));
typedef int      v4i   __attribute__((ext_vector_type(4)));

#ifndef USE_TDM
#define USE_TDM 1
#endif
#if USE_TDM && __has_builtin(__builtin_amdgcn_tensor_load_to_lds)
#define TDM_PATH 1
#else
#define TDM_PATH 0
#endif
#if __has_include(<hip/amd_detail/amd_gfx1250_TDM.h>)
#define TDM_SIX_ARG 1   // therock-10.0 headers -> 6-arg builtin
#else
#define TDM_SIX_ARG 0   // ROCm 7.2 -> 5-arg builtin
#endif

__device__ __forceinline__ v8f wmma_f16(v16h a, v16h b, v8f c) {
  return __builtin_amdgcn_wmma_f32_16x16x32_f16(false, a, false, b, (short)0, c, false, false);
}
__device__ __forceinline__ _Float16 c16(float x) { return (_Float16)x; }
__device__ __forceinline__ unsigned int pk2(float a, float b) {
  union { fp16x2 h; unsigned int u; } c;
  c.h = __builtin_amdgcn_cvt_pkrtz(a, b);   // v_cvt_pk_rtz_f16_f32
  return c.u;
}

// ---------------------------------------------------------------------------
// Generic WMMA GEMM: out(MxN) = epilogue( A(MxK,f32) * W(KxN,f32) )
// Tile: 64(M) x 128(N), BK=32. 256 threads = 8 waves; wave computes 16x64.
// A staged f32->f16 (packed b128 LDS stores); W staged raw f32 via TDM
// (tensor_load_to_lds + s_wait_tensorcnt), converted at fragment build.
// EPI: 0=none 1=+bias 2=+residual(MxN) 3=gelu(x+bias)
// ---------------------------------------------------------------------------
template<int EPI, bool OUT16>
__global__ __launch_bounds__(256) void gemm_wmma_kernel(
    const float* __restrict__ A, const float* __restrict__ W,
    const float* __restrict__ aux, void* __restrict__ outv,
    int M, int N, int K)
{
  __shared__ _Float16 sA[64][32];
  __shared__ float    sB[32][128];

  const int tid  = threadIdx.x;
  const int wave = tid >> 5;
  const int lane = tid & 31;
  const int ml   = lane & 15;      // row/col within fragment
  const int hi   = lane >> 4;      // lane-half select

  const int mblock = blockIdx.x * 64;
  const int nblock = blockIdx.y * 128;
  const int mt = (wave & 3) * 16;          // wave's M sub-tile in LDS
  const int ng = (wave >> 2) * 64;         // wave's N group (0 or 64)

  const v8f vzero = {0.f,0.f,0.f,0.f,0.f,0.f,0.f,0.f};
  v8f acc[4] = {vzero, vzero, vzero, vzero};

  const int ar = tid >> 2;                 // 0..63
  const int ac = (tid & 3) * 8;            // 0,8,16,24
#if !TDM_PATH
  const int br = tid >> 3;                 // 0..31
  const int bc = (tid & 7) * 16;           // 0..112
#endif

#if TDM_PATH
  const unsigned ldsB_off = (unsigned)(uintptr_t)(void*)&sB[0][0];
#endif

  for (int k0 = 0; k0 < K; k0 += 32) {
    // ---- stage A tile (64x32), f32 -> f16, one ds_store_b128 per thread
    {
      const int arow = mblock + ar;
      v4u pa4;
      if (arow < M) {
        const float4* ap = reinterpret_cast<const float4*>(A + (size_t)arow * K + k0 + ac);
        float4 x0 = ap[0], x1 = ap[1];
        pa4.x = pk2(x0.x, x0.y); pa4.y = pk2(x0.z, x0.w);
        pa4.z = pk2(x1.x, x1.y); pa4.w = pk2(x1.z, x1.w);
        if (k0 + 32 < K)   // -> global_prefetch_b8
          __builtin_prefetch(A + (size_t)arow * K + k0 + 32 + ac, 0, 1);
      } else {
        pa4.x = 0u; pa4.y = 0u; pa4.z = 0u; pa4.w = 0u;
      }
      *reinterpret_cast<v4u*>(&sA[ar][ac]) = pa4;
    }

    // ---- stage W tile (32x128 f32) into LDS
#if TDM_PATH
    if (wave == 0) {
      const unsigned long long ga =
          (unsigned long long)(uintptr_t)(W + (size_t)k0 * N + nblock);
      const unsigned rows_left = (unsigned)(K - k0);
      v4u g0;
      g0[0] = 1u;                                   // count=1, user descriptor
      g0[1] = ldsB_off;                             // lds_addr (bytes)
      g0[2] = (unsigned)(ga & 0xffffffffu);         // global_addr[31:0]
      g0[3] = (unsigned)((ga >> 32) & 0x01ffffffu)  // global_addr[56:32]
              | (2u << 30);                         // type = 2 ("image")
      v8i g1;
      g1[0] = (int)(2u << 16);                      // data_size = 4B
      g1[1] = (int)(((unsigned)N & 0xffffu) << 16); // tensor_dim0[15:0]
      g1[2] = (int)((((unsigned)N >> 16) & 0xffffu)
              | ((rows_left & 0xffffu) << 16));     // dim0[31:16] | dim1[15:0]
      g1[3] = (int)(((rows_left >> 16) & 0xffffu)
              | (128u << 16));                      // dim1[31:16] | tile_dim0=128
      g1[4] = 32;                                   // tile_dim1=32, tile_dim2=0
      g1[5] = N;                                    // tensor_dim0_stride[31:0]
      g1[6] = 0;
      g1[7] = 0;
      v4i g2 = {0, 0, 0, 0};
      v4i g3 = {0, 0, 0, 0};
#if TDM_SIX_ARG
      v8i g4 = {0, 0, 0, 0, 0, 0, 0, 0};
      __builtin_amdgcn_tensor_load_to_lds(g0, g1, g2, g3, g4, 0);
#else
      __builtin_amdgcn_tensor_load_to_lds(g0, g1, g2, g3, 0);
#endif
      __builtin_amdgcn_s_wait_tensorcnt(0);
    }
#else
    {
      const float* wrow = W + (size_t)(k0 + br) * N + nblock + bc;
      const float4* wp = reinterpret_cast<const float4*>(wrow);
      #pragma unroll
      for (int j = 0; j < 4; ++j)
        *reinterpret_cast<float4*>(&sB[br][bc + j * 4]) = wp[j];
    }
#endif
    __syncthreads();

    // ---- A fragment (16x32 f16)
    v16h af;
    #pragma unroll
    for (int i = 0; i < 8; ++i) {
      af[i]     = sA[mt + ml][hi*8 + i];
      af[i + 8] = sA[mt + ml][16 + hi*8 + i];
    }
    // ---- 4 B fragments (f32 LDS -> f16) + 4 WMMA
    #pragma unroll
    for (int t = 0; t < 4; ++t) {
      v16h bf;
      #pragma unroll
      for (int i = 0; i < 16; ++i) bf[i] = c16(sB[(i < 8 ? hi*8 + i : 8 + hi*8 + i)][ng + t*16 + ml]);
      acc[t] = wmma_f16(af, bf, acc[t]);
    }
    __syncthreads();
  }

  // ---- epilogue + store (C layout: row = v + hi*8, col = ml)
  #pragma unroll
  for (int t = 0; t < 4; ++t) {
    const int col = nblock + ng + t*16 + ml;
    #pragma unroll
    for (int v = 0; v < 8; ++v) {
      const int row = mblock + mt + hi*8 + v;
      if (row < M) {
        float x = acc[t][v];
        if (EPI == 1) x += aux[col];
        if (EPI == 2) x += aux[(size_t)row * N + col];
        if (EPI == 3) {
          x += aux[col];
          x = 0.5f * x * (1.0f + erff(x * 0.70710678118654752f));
        }
        if (OUT16) ((_Float16*)outv)[(size_t)row * N + col] = (_Float16)x;
        else       ((float*)outv)[(size_t)row * N + col] = x;
      }
    }
  }
}

// ---------------------------------------------------------------------------
// Flash-style WMMA attention. Grid = B*H*(S/64); 128 threads = 4 waves,
// each wave owns one 16-row Q tile. GQA: head h uses kv head h/2.
// ---------------------------------------------------------------------------
__global__ __launch_bounds__(128) void attn_wmma_kernel(
    const _Float16* __restrict__ qb, const _Float16* __restrict__ kb,
    const _Float16* __restrict__ vb, float* __restrict__ ob,
    const int* __restrict__ spd, const int* __restrict__ tlen,
    const float* __restrict__ spd_emb)
{
  __shared__ _Float16 pl[4][16][32];

  const int wave = threadIdx.x >> 5;
  const int lane = threadIdx.x & 31;
  const int ml   = lane & 15;
  const int hi   = lane >> 4;

  const int bid  = blockIdx.x;
  const int qblk = bid & 7;              // S/64 = 8
  const int head = (bid >> 3) & 15;
  const int b    = bid >> 7;
  const int qbase = qblk * 64 + wave * 16;
  const int kvh   = head >> 1;

  const int limit = tlen[b] + 1 + CG;

  // Q A-fragments over HD=64 (two 16x32 fragments), scaled by 1/sqrt(64)
  v16h qa0, qa1;
  {
    const _Float16* qp = qb + ((size_t)(b * CS + qbase + ml)) * (CH*CHD) + head * CHD;
    #pragma unroll
    for (int i = 0; i < 8; ++i) {
      qa0[i]     = c16((float)qp[hi*8 + i] * 0.125f);
      qa0[i + 8] = c16((float)qp[16 + hi*8 + i] * 0.125f);
      qa1[i]     = c16((float)qp[32 + hi*8 + i] * 0.125f);
      qa1[i + 8] = c16((float)qp[48 + hi*8 + i] * 0.125f);
    }
  }

  const v8f vzero = {0.f,0.f,0.f,0.f,0.f,0.f,0.f,0.f};
  v8f oacc[4] = {vzero, vzero, vzero, vzero};
  float mrow[8], lrow[8];
  #pragma unroll
  for (int v = 0; v < 8; ++v) { mrow[v] = -INFINITY; lrow[v] = 0.f; }

  const int* spdb = spd + (size_t)b * 510 * 510;

  for (int it = 0; it < CS / 32; ++it) {
    const int k0 = it * 32;
    v8f s0 = vzero, s1 = vzero;

    // ---- scores: two 16x16 C tiles over HD=64 (2 WMMA each)
    #pragma unroll
    for (int tile = 0; tile < 2; ++tile) {
      const int kc = k0 + tile * 16 + ml;                // key position (B col)
      const _Float16* kp = kb + ((size_t)(b * CS + kc)) * (CHK*CHD) + kvh * CHD;
      v16h bf0, bf1;
      #pragma unroll
      for (int i = 0; i < 16; ++i) {
        bf0[i] = kp[(i < 8 ? hi*8 + i : 8 + hi*8 + i)];          // HD dims 0..31
        bf1[i] = kp[32 + (i < 8 ? hi*8 + i : 8 + hi*8 + i)];     // HD dims 32..63
      }
      if (tile == 0) { s0 = wmma_f16(qa0, bf0, s0); s0 = wmma_f16(qa1, bf1, s0); }
      else           { s1 = wmma_f16(qa0, bf0, s1); s1 = wmma_f16(qa1, bf1, s1); }
    }

    // ---- bias + mask then online softmax
    #pragma unroll
    for (int v = 0; v < 8; ++v) {
      const int qg = qbase + hi*8 + v;
      {
        const int kg = k0 + ml;
        int fs = (qg >= 2 && kg >= 2) ? spdb[(qg - 2) * 510 + (kg - 2)]
                                      : ((qg == kg) ? 0 : CMAXSPD + 1);
        float bias = spd_emb[fs * CH + head];
        s0[v] += (kg < limit) ? bias : -INFINITY;
      }
      {
        const int kg = k0 + 16 + ml;
        int fs = (qg >= 2 && kg >= 2) ? spdb[(qg - 2) * 510 + (kg - 2)]
                                      : ((qg == kg) ? 0 : CMAXSPD + 1);
        float bias = spd_emb[fs * CH + head];
        s1[v] += (kg < limit) ? bias : -INFINITY;
      }
    }

    #pragma unroll
    for (int v = 0; v < 8; ++v) {
      float tmax = fmaxf(s0[v], s1[v]);
      #pragma unroll
      for (int m = 1; m <= 8; m <<= 1) tmax = fmaxf(tmax, __shfl_xor(tmax, m, 32));
      const float mn  = fmaxf(mrow[v], tmax);
      const float fac = __expf(mrow[v] - mn);
      mrow[v] = mn;
      const float p0 = __expf(s0[v] - mn);
      const float p1 = __expf(s1[v] - mn);
      float rs = p0 + p1;
      #pragma unroll
      for (int m = 1; m <= 8; m <<= 1) rs += __shfl_xor(rs, m, 32);
      lrow[v] = lrow[v] * fac + rs;
      #pragma unroll
      for (int t = 0; t < 4; ++t) oacc[t][v] *= fac;
      pl[wave][hi*8 + v][ml]      = (_Float16)p0;
      pl[wave][hi*8 + v][16 + ml] = (_Float16)p1;
    }

    // ---- P A-fragment (16x32) from LDS
    v16h pa;
    #pragma unroll
    for (int i = 0; i < 8; ++i) {
      pa[i]     = pl[wave][ml][hi*16 + i];
      pa[i + 8] = pl[wave][ml][hi*16 + 8 + i];
    }
    // ---- P * V (4 output tiles over HD)
    #pragma unroll
    for (int t = 0; t < 4; ++t) {
      const _Float16* vp = vb + ((size_t)(b * CS + k0 + hi*16)) * (CHK*CHD)
                              + kvh * CHD + t * 16 + ml;
      v16h bf;
      #pragma unroll
      for (int i = 0; i < 16; ++i) bf[i] = vp[(size_t)i * (CHK*CHD)];
      oacc[t] = wmma_f16(pa, bf, oacc[t]);
    }
  }

  // ---- finalize: divide by l, store fp32
  #pragma unroll
  for (int t = 0; t < 4; ++t) {
    #pragma unroll
    for (int v = 0; v < 8; ++v) {
      const float val = oacc[t][v] / lrow[v];
      ob[((size_t)(b * CS + qbase + hi*8 + v)) * (CH*CHD) + head * CHD + t * 16 + ml] = val;
    }
  }
}

// ---------------------------------------------------------------------------
// RMSNorm: one block per row of 1024
// ---------------------------------------------------------------------------
__global__ __launch_bounds__(256) void rmsnorm_kernel(
    const float* __restrict__ in, const float* __restrict__ w,
    float* __restrict__ out)
{
  __shared__ float red[8];
  const int row = blockIdx.x;
  const int tid = threadIdx.x;
  const float* x = in + (size_t)row * CD;
  float v0 = x[tid], v1 = x[tid + 256], v2 = x[tid + 512], v3 = x[tid + 768];
  float s = v0*v0 + v1*v1 + v2*v2 + v3*v3;
  #pragma unroll
  for (int m = 1; m <= 16; m <<= 1) s += __shfl_xor(s, m, 32);
  if ((tid & 31) == 0) red[tid >> 5] = s;
  __syncthreads();
  float tot = 0.f;
  #pragma unroll
  for (int i = 0; i < 8; ++i) tot += red[i];
  const float r = rsqrtf(tot * (1.0f / CD) + 1e-6f);
  float* o = out + (size_t)row * CD;
  o[tid]       = v0 * r * w[tid];
  o[tid + 256] = v1 * r * w[tid + 256];
  o[tid + 512] = v2 * r * w[tid + 512];
  o[tid + 768] = v3 * r * w[tid + 768];
}

// ---------------------------------------------------------------------------
// Assemble h[B,S,D] = (graph|sos|in_proj) + type_emb + ne + sinusoid PE
// ---------------------------------------------------------------------------
__global__ __launch_bounds__(256) void build_h_kernel(
    const float* __restrict__ gf, const float* __restrict__ hdata,
    const float* __restrict__ sos, const float* __restrict__ ne,
    const float* __restrict__ type_emb, const int* __restrict__ subids,
    float* __restrict__ h)
{
  const size_t idx = (size_t)blockIdx.x * 256 + threadIdx.x;   // over B*S*D
  const int c = (int)(idx & (CD - 1));
  const size_t rs = idx >> 10;
  const int s = (int)(rs & (CS - 1));
  const int b = (int)(rs >> 9);

  const float i2  = (float)(c & ~1);
  const float div = __expf(i2 * (-9.210340371976184f / (float)CD));
  const float ang = (float)s * div;
  const float pe  = (c & 1) ? __cosf(ang) : __sinf(ang);

  float val;
  if (s < CG) {
    val = gf[((size_t)b * CG + s) * CD + c] + type_emb[CD + c];
  } else {
    const int t = s - CG;
    float base; int sub;
    if (t == 0) { base = sos[c]; sub = 0; }
    else {
      base = hdata[((size_t)b * CT + (t - 1)) * CD + c];
      sub  = subids[b * CT + (t - 1)];
    }
    val = base + type_emb[c] + ne[(size_t)sub * CD + c];
  }
  h[idx] = val + pe;
}

__global__ __launch_bounds__(256) void silumul_kernel(
    float* __restrict__ u, const float* __restrict__ g, size_t n)
{
  const size_t i = (size_t)blockIdx.x * 256 + threadIdx.x;
  if (i < n) {
    const float x = u[i];
    u[i] = (x / (1.0f + __expf(-x))) * g[i];
  }
}

__global__ __launch_bounds__(256) void gather_seq_kernel(
    const float* __restrict__ xn, float* __restrict__ hseq)
{
  const size_t idx = (size_t)blockIdx.x * 256 + threadIdx.x;  // over 1792*1024
  const int c = (int)(idx & (CD - 1));
  const int row = (int)(idx >> 10);
  const int b = row / (CT + 1);
  const int t = row % (CT + 1);
  hseq[idx] = xn[((size_t)b * CS + CG + t) * CD + c];
}

// ---------------------------------------------------------------------------
// Host-side GEMM dispatch
// ---------------------------------------------------------------------------
static inline void gemm(hipStream_t st, const float* A, const float* W,
                        const float* aux, void* out, int M, int N, int K,
                        int epi, bool out16)
{
  dim3 grid((M + 63) / 64, N / 128), blk(256);
  if (out16) {
    gemm_wmma_kernel<0, true><<<grid, blk, 0, st>>>(A, W, aux, out, M, N, K);
    return;
  }
  switch (epi) {
    case 0: gemm_wmma_kernel<0, false><<<grid, blk, 0, st>>>(A, W, aux, out, M, N, K); break;
    case 1: gemm_wmma_kernel<1, false><<<grid, blk, 0, st>>>(A, W, aux, out, M, N, K); break;
    case 2: gemm_wmma_kernel<2, false><<<grid, blk, 0, st>>>(A, W, aux, out, M, N, K); break;
    default: gemm_wmma_kernel<3, false><<<grid, blk, 0, st>>>(A, W, aux, out, M, N, K); break;
  }
}

extern "C" void kernel_launch(void* const* d_in, const int* in_sizes, int n_in,
                              void* d_out, int out_size, void* d_ws, size_t ws_size,
                              hipStream_t stream)
{
  const float* graph     = (const float*)d_in[0];
  const float* inputf    = (const float*)d_in[1];
  const float* in_proj_w = (const float*)d_in[2];
  const float* in_proj_b = (const float*)d_in[3];
  const float* sos       = (const float*)d_in[4];
  const float* ne        = (const float*)d_in[5];
  const float* type_emb  = (const float*)d_in[6];
  const float* spd_emb   = (const float*)d_in[7];
  const float* wq        = (const float*)d_in[8];
  const float* wk        = (const float*)d_in[9];
  const float* wv        = (const float*)d_in[10];
  const float* wo        = (const float*)d_in[11];
  const float* an_w      = (const float*)d_in[12];
  const float* fn_w      = (const float*)d_in[13];
  const float* w1        = (const float*)d_in[14];
  const float* w2        = (const float*)d_in[15];
  const float* w3        = (const float*)d_in[16];
  const float* norm_w    = (const float*)d_in[17];
  const float* out1_w    = (const float*)d_in[18];
  const float* out1_b    = (const float*)d_in[19];
  const float* out2_w    = (const float*)d_in[20];
  const float* out2_b    = (const float*)d_in[21];
  const int*   subids    = (const int*)d_in[22];
  const int*   tlen      = (const int*)d_in[23];
  const int*   spd       = (const int*)d_in[24];
  float* out = (float*)d_out;

  char* ws = (char*)d_ws;
  const size_t ROWS = (size_t)CB * CS;          // 2048
  const size_t OFF_H  = 0;                                       // f32 2048x1024
  const size_t OFF_X  = OFF_H  + ROWS * CD * 4;                  // f32 2048x1024
  const size_t OFF_Q  = OFF_X  + ROWS * CD * 4;                  // f16 2048x1024
  const size_t OFF_K  = OFF_Q  + ROWS * CD * 2;                  // f16 2048x512
  const size_t OFF_V  = OFF_K  + ROWS * (CHK*CHD) * 2;           // f16 2048x512
  const size_t OFF_O  = OFF_V  + ROWS * (CHK*CHD) * 2;           // f32 2048x1024
  const size_t OFF_U  = OFF_O  + ROWS * CD * 4;                  // f32 2048x2816
  const size_t OFF_G3 = OFF_U  + ROWS * CHID * 4;                // f32 2048x2816

  float* h    = (float*)(ws + OFF_H);
  float* x    = (float*)(ws + OFF_X);
  _Float16* qb = (_Float16*)(ws + OFF_Q);
  _Float16* kb = (_Float16*)(ws + OFF_K);
  _Float16* vb = (_Float16*)(ws + OFF_V);
  float* ob   = (float*)(ws + OFF_O);
  float* u    = (float*)(ws + OFF_U);
  float* g3   = (float*)(ws + OFF_G3);
  float* hseq = (float*)(ws + OFF_U);    // reuse after layers
  float* g1   = (float*)(ws + OFF_G3);   // reuse after layers

  // 1) in_proj: (B*T, FD) @ (FD, D) + b  -> ob
  gemm(stream, inputf, in_proj_w, in_proj_b, ob, CB * CT, CD, CFD, 1, false);

  // 2) assemble h
  build_h_kernel<<<(CB * CS * CD) / 256, 256, 0, stream>>>(
      graph, ob, sos, ne, type_emb, subids, h);

  // 3) transformer layers
  for (int l = 0; l < CL; ++l) {
    const float* wq_ = wq + (size_t)l * CD * (CH*CHD);
    const float* wk_ = wk + (size_t)l * CD * (CHK*CHD);
    const float* wv_ = wv + (size_t)l * CD * (CHK*CHD);
    const float* wo_ = wo + (size_t)l * (CH*CHD) * CD;
    const float* w1_ = w1 + (size_t)l * CD * CHID;
    const float* w2_ = w2 + (size_t)l * CHID * CD;
    const float* w3_ = w3 + (size_t)l * CD * CHID;

    rmsnorm_kernel<<<(int)ROWS, 256, 0, stream>>>(h, an_w + (size_t)l * CD, x);
    gemm(stream, x, wq_, nullptr, qb, (int)ROWS, CH*CHD,  CD, 0, true);
    gemm(stream, x, wk_, nullptr, kb, (int)ROWS, CHK*CHD, CD, 0, true);
    gemm(stream, x, wv_, nullptr, vb, (int)ROWS, CHK*CHD, CD, 0, true);

    attn_wmma_kernel<<<CB * CH * (CS / 64), 128, 0, stream>>>(
        qb, kb, vb, ob, spd, tlen, spd_emb);

    gemm(stream, ob, wo_, h, h, (int)ROWS, CD, CH*CHD, 2, false);  // h += o @ wo

    rmsnorm_kernel<<<(int)ROWS, 256, 0, stream>>>(h, fn_w + (size_t)l * CD, x);
    gemm(stream, x, w1_, nullptr, u,  (int)ROWS, CHID, CD, 0, false);
    gemm(stream, x, w3_, nullptr, g3, (int)ROWS, CHID, CD, 0, false);
    silumul_kernel<<<(int)((ROWS * CHID + 255) / 256), 256, 0, stream>>>(
        u, g3, ROWS * CHID);
    gemm(stream, u, w2_, h, h, (int)ROWS, CD, CHID, 2, false);     // h += act @ w2
  }

  // 4) final norm + head
  rmsnorm_kernel<<<(int)ROWS, 256, 0, stream>>>(h, norm_w, x);
  const int SEQR = CB * (CT + 1);   // 1792
  gather_seq_kernel<<<(SEQR * CD) / 256, 256, 0, stream>>>(x, hseq);
  gemm(stream, hseq, out1_w, out1_b, g1, SEQR, CD, CD, 3, false);  // gelu(x@W1+b)
  gemm(stream, g1, out2_w, out2_b, out, SEQR, CFD, CD, 1, false);  // @W2+b -> d_out
}
